// NoisyTopKRouter_44985487458588
// MI455X (gfx1250) — matrix operand order repaired and checked by reading, not compiled, not run on hardware
//
#include <hip/hip_runtime.h>
#include <hip/hip_bf16.h>

#define BB 4
#define SS 4096
#define DD 2048
#define EE 64
#define KK 8
#define NROWS (BB * SS)

typedef __attribute__((ext_vector_type(16))) __bf16 v16bf;
typedef __attribute__((ext_vector_type(8)))  float  v8f;

__device__ __forceinline__ unsigned int bf16_rne(unsigned int x) {
  return (x + 0x7FFFu + ((x >> 16) & 1u)) >> 16;
}

// Pack two f32 -> one dword of two bf16 (lowers to v_cvt_pk_bf16_f32 where available)
__device__ __forceinline__ unsigned int pack_bf16(float lo, float hi) {
  union { __hip_bfloat162 h; unsigned int u; } c;
  c.h = __float22bfloat162_rn(make_float2(lo, hi));
  return c.u;
}

// Convert W [D][E] f32 -> bf16 transposed [E][D] so B-fragments are contiguous.
__global__ __launch_bounds__(256) void wt_convert_tr(const float* __restrict__ w,
                                                     unsigned short* __restrict__ out) {
  int t = blockIdx.x * 256 + threadIdx.x;   // t in [0, EE*DD)
  int e = t >> 11;                          // / DD
  int d = t & (DD - 1);
  out[t] = (unsigned short)bf16_rne(__float_as_uint(w[(size_t)d * EE + e]));
}

__global__ __launch_bounds__(256) void router_main(
    const float* __restrict__ hidden, const float* __restrict__ eps,
    const unsigned short* __restrict__ wtR, const unsigned short* __restrict__ wtN,
    const float* __restrict__ bR, const float* __restrict__ bN,
    float* __restrict__ probs, int* __restrict__ topk) {
  // Partial accumulators: [kh][mat][row(32)][expert(64)]  = 32 KB
  __shared__ float s_part[2][2][32][EE];
  // Noisy logits for this block's 32 rows                 = 8 KB
  __shared__ float s_noisy[32][EE];

  const int wave = threadIdx.x >> 5;
  const int lane = threadIdx.x & 31;
  const int l15  = lane & 15;
  const int hi   = lane >> 4;           // 0 or 1
  const int mh   = wave & 1;            // which 16-row half
  const int mat  = (wave >> 1) & 1;     // 0 = route, 1 = noise
  const int kh   = wave >> 2;           // which K half
  const int blockRow = blockIdx.x * 32;
  const int rowTile  = blockRow + mh * 16;

  const unsigned short* __restrict__ wt = mat ? wtN : wtR;

  v8f acc[4] = {};

  // A layout (16-bit A 16x32): lanes 0-15: M=lane, K chunks [0,8)+[16,24);
  //                            lanes 16-31: M=lane-16, K chunks [8,16)+[24,32).
  const float* hrow = hidden + (size_t)(rowTile + l15) * DD;
  const int ka0 = hi ? 8 : 0;
  const int ka1 = ka0 + 16;
  const int kbeg = kh * (DD / 2);
  const int kend = kbeg + (DD / 2);

  for (int kc = kbeg; kc < kend; kc += 32) {
    // ---- issue all global loads first so they clause together ----
    const float4 f0 = *(const float4*)(hrow + kc + ka0);
    const float4 f1 = *(const float4*)(hrow + kc + ka0 + 4);
    const float4 f2 = *(const float4*)(hrow + kc + ka1);
    const float4 f3 = *(const float4*)(hrow + kc + ka1 + 4);

    // B layout (32x16 bf16): lanes 0-15: N=lane, K chunks [0,8)+[16,24); lanes 16-31: N=lane-16, K chunks [8,16)+[24,32).
    uint4 bq[8];
#pragma unroll
    for (int nb = 0; nb < 4; ++nb) {
      const unsigned short* wp = wt + (size_t)(nb * 16 + l15) * DD + kc + hi * 8;
      bq[2 * nb]     = *(const uint4*)(wp);
      bq[2 * nb + 1] = *(const uint4*)(wp + 16);
    }

    union { v16bf v; unsigned int u[8]; } A;
    A.u[0] = pack_bf16(f0.x, f0.y); A.u[1] = pack_bf16(f0.z, f0.w);
    A.u[2] = pack_bf16(f1.x, f1.y); A.u[3] = pack_bf16(f1.z, f1.w);
    A.u[4] = pack_bf16(f2.x, f2.y); A.u[5] = pack_bf16(f2.z, f2.w);
    A.u[6] = pack_bf16(f3.x, f3.y); A.u[7] = pack_bf16(f3.z, f3.w);

#pragma unroll
    for (int nb = 0; nb < 4; ++nb) {
      union { v16bf v; uint4 q[2]; } Bf;
      Bf.q[0] = bq[2 * nb];
      Bf.q[1] = bq[2 * nb + 1];
      acc[nb] = __builtin_amdgcn_wmma_f32_16x16x32_bf16(
          false, A.v, false, Bf.v, (short)0, acc[nb], false, false);
    }
  }

  // Store partials in WMMA C/D layout:
  // VGPR r, lanes 0-15: M=r, N=lane; lanes 16-31: M=8+r, N=lane-16.
#pragma unroll
  for (int nb = 0; nb < 4; ++nb) {
    const int e = nb * 16 + l15;
#pragma unroll
    for (int r = 0; r < 8; ++r) {
      s_part[kh][mat][mh * 16 + r + hi * 8][e] = acc[nb][r];
    }
  }
  __syncthreads();

  // ---- combine partials + bias + softplus + noisy logits (flat over 32x64) ----
  {
    const int row = threadIdx.x >> 3;          // 0..31
    const int e0  = (threadIdx.x & 7) * 8;     // 0,8,...,56
    const float* ep = eps + ((size_t)blockRow + row) * EE + e0;
#pragma unroll
    for (int j = 0; j < 8; ++j) {
      const int e = e0 + j;
      const float lg = s_part[0][0][row][e] + s_part[1][0][row][e] + bR[e];
      const float x  = s_part[0][1][row][e] + s_part[1][1][row][e] + bN[e];
      const float sp = fmaxf(x, 0.0f) + log1pf(__expf(-fabsf(x)));  // stable softplus
      s_noisy[row][e] = lg + ep[j] * sp;
    }
  }
  __syncthreads();

  // ---- per-row top-8 + softmax + scatter: threads 0..31 each own one row ----
  if (threadIdx.x < 32) {
    const int row = threadIdx.x;
    const size_t grow = (size_t)blockRow + row;
    float v[KK];
    int   id[KK];
#pragma unroll
    for (int j = 0; j < KK; ++j) { v[j] = -3.4e38f; id[j] = 0; }

    for (int e = 0; e < EE; ++e) {
      const float nz = s_noisy[row][e];
      if (nz > v[KK - 1]) {                 // strict > keeps earlier index on ties
        v[KK - 1] = nz; id[KK - 1] = e;
#pragma unroll
        for (int j = KK - 1; j > 0; --j) {
          if (v[j] > v[j - 1]) {
            const float tv = v[j]; v[j] = v[j - 1]; v[j - 1] = tv;
            const int   ti = id[j]; id[j] = id[j - 1]; id[j - 1] = ti;
          }
        }
      }
    }

    const float m = v[0];
    float p[KK], sum = 0.0f;
#pragma unroll
    for (int j = 0; j < KK; ++j) { p[j] = __expf(v[j] - m); sum += p[j]; }
    const float inv = 1.0f / sum;

    // Stage dense 64-wide prob row in LDS (reuse s_noisy row), then vector-store.
    float* srow = &s_noisy[row][0];
#pragma unroll
    for (int e = 0; e < EE; ++e) srow[e] = 0.0f;
#pragma unroll
    for (int j = 0; j < KK; ++j) srow[id[j]] = p[j] * inv;

    float* op = probs + grow * EE;
#pragma unroll
    for (int e = 0; e < EE; e += 4) *(float4*)(op + e) = *(const float4*)(srow + e);

    int* ti = topk + grow * KK;
#pragma unroll
    for (int j = 0; j < KK; ++j) ti[j] = id[j];
  }
}

extern "C" void kernel_launch(void* const* d_in, const int* in_sizes, int n_in,
                              void* d_out, int out_size, void* d_ws, size_t ws_size,
                              hipStream_t stream) {
  const float* hidden = (const float*)d_in[0];  // [B,S,D]
  const float* eps    = (const float*)d_in[1];  // [B,S,E]
  const float* Wr     = (const float*)d_in[2];  // [D,E]
  const float* br     = (const float*)d_in[3];  // [E]
  const float* Wn     = (const float*)d_in[4];  // [D,E]
  const float* bn     = (const float*)d_in[5];  // [E]

  unsigned short* wtR = (unsigned short*)d_ws;            // [E][D] bf16
  unsigned short* wtN = wtR + (size_t)EE * DD;            // [E][D] bf16 (512 KB total)

  float* probs = (float*)d_out;                               // [B,S,E] f32
  int*   topk  = (int*)((float*)d_out + (size_t)NROWS * EE);  // [B,S,K] i32

  dim3 cgrid((EE * DD) / 256);
  wt_convert_tr<<<cgrid, 256, 0, stream>>>(Wr, wtR);
  wt_convert_tr<<<cgrid, 256, 0, stream>>>(Wn, wtN);

  // 512 blocks x 8 waves = 4096 waves: (row-half, matrix, K-half) split per 32-row tile.
  router_main<<<dim3(NROWS / 32), 256, 0, stream>>>(hidden, eps, wtR, wtN, br, bn,
                                                    probs, topk);
}
